// DeformScaledDotAttnLayer_22222160789700
// MI455X (gfx1250) — matrix operand order repaired and checked by reading, not compiled, not run on hardware
//
#include <hip/hip_runtime.h>
#include <math.h>
#include <stdint.h>

// ---------------------------------------------------------------------------
// CDNA5 (gfx1250) deformable scaled-dot attention.
// wave32, WMMA f16->f32, double-buffered async global->LDS staging,
// channel-last activations.
// ---------------------------------------------------------------------------

typedef __attribute__((ext_vector_type(16))) _Float16 v16h;
typedef __attribute__((ext_vector_type(8)))  float    v8f;

#define BATCH 4
#define DIM   256
#define HEADS 8
#define DPH   32
#define HH    56
#define WW    56
#define QP    3136      // 56*56
#define NH    28
#define NP    784       // 28*28
#define KPAD  832       // 13*64 padded key/value pixels
#define EPSLN 1e-5f
#define W1ELEMS (25 * 256 * 512)

__device__ __forceinline__ v8f wmma32(v16h a, v16h b, v8f c) {
  return __builtin_amdgcn_wmma_f32_16x16x32_f16(
      false, a, false, b, (short)0, c, false, false);
}

// A fragment (16x32 f16), row-major source, stride lda elements.
template <typename T>
__device__ __forceinline__ v16h load_a(const T* p, int lda) {
  int lane = threadIdx.x & 31;
  const T* r = p + (lane & 15) * lda + ((lane < 16) ? 0 : 8);
  v16h a;
#pragma unroll
  for (int i = 0; i < 8; ++i) a[i] = (_Float16)r[i];
#pragma unroll
  for (int i = 0; i < 8; ++i) a[8 + i] = (_Float16)r[16 + i];
  return a;
}

// B fragment (32x16 f16) from an N-major tile: p[n*ldk + k] == B[k][n].
__device__ __forceinline__ v16h load_bt(const _Float16* p, int ldk) {
  int lane = threadIdx.x & 31;
  const _Float16* q = p + (lane & 15) * ldk + ((lane < 16) ? 0 : 8);
  v16h b;
#pragma unroll
  for (int i = 0; i < 8; ++i) b[i] = q[i];
#pragma unroll
  for (int i = 0; i < 8; ++i) b[8 + i] = q[16 + i];
  return b;
}

// async 16-byte global->LDS copy (ASYNCcnt-tracked, CDNA5 §15.18 op 98)
__device__ __forceinline__ void async_b128(uint32_t lds_off, const void* g) {
  asm volatile("global_load_async_to_lds_b128 %0, %1, off"
               :: "v"(lds_off), "v"(g) : "memory");
}
__device__ __forceinline__ void wait_async0() {
  asm volatile("s_wait_asynccnt 0" ::: "memory");
}
__device__ __forceinline__ void wait_async1() {   // <=1 outstanding
  asm volatile("s_wait_asynccnt 1" ::: "memory");
}
__device__ __forceinline__ void wait_async4() {   // <=4 outstanding
  asm volatile("s_wait_asynccnt 4" ::: "memory");
}

// C fragment mapping: col = lane&15, row = vgpr + (lane<16 ? 0 : 8).

// ---------------------------------------------------------------------------
// K0: weight pre-layout to f16
// ---------------------------------------------------------------------------
__global__ void convert_weights(const float* __restrict__ so_w1,
                                const float* __restrict__ q_w,
                                const float* __restrict__ o_w,
                                _Float16* __restrict__ w1h,
                                _Float16* __restrict__ qwh,
                                _Float16* __restrict__ owh) {
  int idx = blockIdx.x * 256 + threadIdx.x;
  if (idx < W1ELEMS) {
    int ic = idx & 511, oc = (idx >> 9) & 255, tap = idx >> 17;
    w1h[idx] = (_Float16)so_w1[(oc * 512 + ic) * 25 + tap];
  } else if (idx < W1ELEMS + 65536) {
    int e = idx - W1ELEMS;
    qwh[e] = (_Float16)q_w[e];
  } else if (idx < W1ELEMS + 131072) {
    int e = idx - (W1ELEMS + 65536);
    owh[e] = (_Float16)o_w[e];
  }
}

// ---------------------------------------------------------------------------
// K1: 2x2 avg pool + concat + zero-pad 2, CHANNEL-LAST: recP[b][32y][32x][512c]
// ---------------------------------------------------------------------------
__global__ void pool_pad_kernel(const float* __restrict__ q,
                                const float* __restrict__ xin,
                                _Float16* __restrict__ recP) {
  int idx = blockIdx.x * blockDim.x + threadIdx.x;
  if (idx >= BATCH * 512 * 32 * 32) return;
  int c  = idx & 511;
  int xx = (idx >> 9) & 31;
  int yy = (idx >> 14) & 31;
  int b  = idx >> 19;
  float v = 0.f;
  if (xx >= 2 && xx < 30 && yy >= 2 && yy < 30) {
    int py = yy - 2, px = xx - 2;
    const float* src = (c < DIM) ? q : xin;
    int cc = c & (DIM - 1);
    const float* p = src + ((b * DIM + cc) * HH + py * 2) * WW + px * 2;
    v = 0.25f * (p[0] + p[1] + p[WW] + p[WW + 1]);
  }
  recP[idx] = (_Float16)v;
}

// ---------------------------------------------------------------------------
// K2: fused offset head: 5x5 conv (implicit GEMM) + LN + GELU + 1x1 + tanh
//     grid (49, B), 256 threads (8 waves x 2 M-tiles).
//     Double-buffered 4KB async B stages; 4 WMMA steps per stage.
// ---------------------------------------------------------------------------
__global__ void __launch_bounds__(256)
offsets_kernel(const _Float16* __restrict__ recP,
               const _Float16* __restrict__ w1h,
               const float* __restrict__ so_b1,
               const float* __restrict__ ln_g,  const float* __restrict__ ln_b,
               const float* __restrict__ so_w2, const float* __restrict__ so_b2,
               float* __restrict__ offb) {
  __shared__ __align__(16) _Float16 sB[2][16 * 128];  // [16 px][128 ic] x2
  __shared__ float    sH[256 * 16];                   // conv out [256 oc][16 px]
  __shared__ __align__(16) _Float16 sHh[16 * 256];    // post-GELU [16 px][256]
  __shared__ float    sRed[256];
  __shared__ float    sMu[16], sInv[16];

  const int b    = blockIdx.y;
  const int p0   = blockIdx.x * 16;
  const int tid  = threadIdx.x;
  const int wave = tid >> 5;
  const int lane = tid & 31;

  // async staging geometry: every thread issues exactly one b128 per stage
  const int sj = tid >> 4;          // pixel 0..15
  const int sq = tid & 15;          // 8-channel quarter 0..15
  const int sp = p0 + sj;
  const int spy = sp / NH, spx = sp % NH;
  const _Float16* gpix =
      recP + (((size_t)b * 32 + spy) * 32 + spx) * 512 + sq * 8;
  const uint32_t lds0 =
      (uint32_t)(uintptr_t)&sB[0][0] + (uint32_t)(sj * 128 + sq * 8) * 2;
  const uint32_t lds1 =
      (uint32_t)(uintptr_t)&sB[1][0] + (uint32_t)(sj * 128 + sq * 8) * 2;

  v8f acc0 = {}, acc1 = {};

  // flattened pipeline: s = tap*4 + kc4, 100 stages, double-buffered
  async_b128(lds0, gpix);   // s=0: tap 0 (ty=tx=0), kc4=0
  for (int s = 0; s < 100; ++s) {
    if (s + 1 < 100) {
      const int sn = s + 1, tapn = sn >> 2;
      const int tyn = tapn / 5, txn = tapn % 5;
      async_b128((sn & 1) ? lds1 : lds0,
                 gpix + (tyn * 32 + txn) * 512 + (sn & 3) * 128);
      wait_async1();   // async loads complete in order -> stage s landed
    } else {
      wait_async0();
    }
    __syncthreads();
    const int tap = s >> 2;
    const _Float16* wbase =
        w1h + ((size_t)tap * 256 + wave * 32) * 512 + (s & 3) * 128;
    const _Float16* bbase = (s & 1) ? &sB[1][0] : &sB[0][0];
#pragma unroll
    for (int sub = 0; sub < 4; ++sub) {
      v16h a0 = load_a(wbase + sub * 32, 512);
      v16h a1 = load_a(wbase + 16 * 512 + sub * 32, 512);
      v16h bf = load_bt(bbase + sub * 32, 128);
      acc0 = wmma32(a0, bf, acc0);
      acc1 = wmma32(a1, bf, acc1);
    }
    __syncthreads();
  }

  const int col = lane & 15;
  const int rb  = (lane < 16) ? 0 : 8;
#pragma unroll
  for (int j = 0; j < 8; ++j) {
    int m0 = (wave * 2 + 0) * 16 + rb + j;
    int m1 = (wave * 2 + 1) * 16 + rb + j;
    sH[m0 * 16 + col] = acc0[j] + so_b1[m0];
    sH[m1 * 16 + col] = acc1[j] + so_b1[m1];
  }
  __syncthreads();

  // channel LayerNorm + exact GELU: 16 threads per pixel, LDS tree reduce
  const int px_ = tid & 15, g_ = tid >> 4;
  {
    float part = 0.f;
#pragma unroll 4
    for (int i = 0; i < 16; ++i) part += sH[(g_ * 16 + i) * 16 + px_];
    sRed[g_ * 16 + px_] = part;
  }
  __syncthreads();
  if (tid < 16) {
    float mu = 0.f;
#pragma unroll 4
    for (int i = 0; i < 16; ++i) mu += sRed[i * 16 + tid];
    sMu[tid] = mu * (1.f / DIM);
  }
  __syncthreads();
  const float mu = sMu[px_];
  {
    float part = 0.f;
#pragma unroll 4
    for (int i = 0; i < 16; ++i) {
      float d = sH[(g_ * 16 + i) * 16 + px_] - mu;
      part += d * d;
    }
    sRed[g_ * 16 + px_] = part;
  }
  __syncthreads();
  if (tid < 16) {
    float v = 0.f;
#pragma unroll 4
    for (int i = 0; i < 16; ++i) v += sRed[i * 16 + tid];
    sInv[tid] = rsqrtf(v * (1.f / DIM) + EPSLN);
  }
  __syncthreads();
  const float inv = sInv[px_];
#pragma unroll 4
  for (int i = 0; i < 16; ++i) {
    int c = g_ * 16 + i;
    float hn = (sH[c * 16 + px_] - mu) * inv * ln_g[c] + ln_b[c];
    float ge = 0.5f * hn * (1.f + erff(hn * 0.70710678118654752f));
    sHh[px_ * 256 + c] = (_Float16)ge;
  }
  __syncthreads();

  // offsets = tanh(so_w2 @ h + so_b2): K=256 -> 8 WMMA on wave 0
  if (wave == 0) {
    v8f oc_ = {};
    for (int kc = 0; kc < 8; ++kc) {
      v16h af = load_a(so_w2 + kc * 32, 256);
      v16h bf = load_bt(sHh + kc * 32, 256);
      oc_ = wmma32(af, bf, oc_);
    }
#pragma unroll
    for (int j = 0; j < 8; ++j) {
      int ch = rb + j;
      offb[(b * 16 + ch) * NP + p0 + col] = tanhf(oc_[j] + so_b2[ch]);
    }
  }
}

// ---------------------------------------------------------------------------
// K3: q projection, output transposed f16 q_t[bh][pix][32].
// ---------------------------------------------------------------------------
__global__ void __launch_bounds__(256)
qproj_kernel(const float* __restrict__ query, const float* __restrict__ pos,
             const _Float16* __restrict__ qwh, const float* __restrict__ q_b,
             _Float16* __restrict__ q_t) {
  __shared__ __align__(16) _Float16 sB[64 * 32];   // [64 px][32 k]

  const int b    = blockIdx.y;
  const int n0   = blockIdx.x * 64;
  const int wave = threadIdx.x >> 5;
  const int lane = threadIdx.x & 31;

  v8f acc[8];
#pragma unroll
  for (int i = 0; i < 8; ++i) acc[i] = (v8f){};

  for (int kc = 0; kc < 8; ++kc) {
#pragma unroll
    for (int it = 0; it < 8; ++it) {
      int e = it * 256 + threadIdx.x;
      int r = e >> 6, j = e & 63;
      int k = kc * 32 + r, pix = n0 + j;
      sB[j * 32 + r] = (_Float16)(query[(b * DIM + k) * QP + pix] +
                                  pos[(b * DIM + k) * QP + pix]);
    }
    __syncthreads();
    v16h a0 = load_a(qwh + (wave * 32 + 0) * DIM + kc * 32, DIM);
    v16h a1 = load_a(qwh + (wave * 32 + 16) * DIM + kc * 32, DIM);
#pragma unroll
    for (int nt = 0; nt < 4; ++nt) {
      v16h bf = load_bt(sB + nt * 16 * 32, 32);
      acc[nt]     = wmma32(a0, bf, acc[nt]);
      acc[4 + nt] = wmma32(a1, bf, acc[4 + nt]);
    }
    __syncthreads();
  }

  const int col = lane & 15;
  const int rb  = (lane < 16) ? 0 : 8;
#pragma unroll
  for (int mt = 0; mt < 2; ++mt)
#pragma unroll
    for (int nt = 0; nt < 4; ++nt)
#pragma unroll
      for (int j = 0; j < 8; ++j) {
        int c   = (wave * 2 + mt) * 16 + rb + j;
        int pix = n0 + nt * 16 + col;
        float v = acc[mt * 4 + nt][j] + q_b[c];
        int bh = b * HEADS + (c >> 5);
        q_t[((size_t)bh * QP + pix) * DPH + (c & 31)] = (_Float16)v;
      }
}

// ---------------------------------------------------------------------------
// K4: bilinear sampling fused with k/v WMMA projections.
//     k -> pixel-major kt[bh][KPAD][32], v -> d-major vd[bh][32][KPAD]
// ---------------------------------------------------------------------------
__global__ void __launch_bounds__(64)
sample_kv_kernel(const float* __restrict__ xin, const float* __restrict__ ref,
                 const float* __restrict__ offb,
                 const float* __restrict__ k_w, const float* __restrict__ k_b,
                 const float* __restrict__ v_w, const float* __restrict__ v_b,
                 _Float16* __restrict__ kt, _Float16* __restrict__ vd) {
  __shared__ float    sLoc[16][2];
  __shared__ __align__(16) _Float16 sS[16 * 32];    // sampled [px][d]
  __shared__ __align__(16) _Float16 sWk[32 * 32];
  __shared__ __align__(16) _Float16 sWv[32 * 32];

  const int bh   = blockIdx.y;
  const int b    = bh >> 3;
  const int head = bh & 7;
  const int p0   = blockIdx.x * 16;
  const int tid  = threadIdx.x;

  for (int e = tid; e < 1024; e += 64) {
    sWk[e] = (_Float16)k_w[e];
    sWv[e] = (_Float16)v_w[e];
  }
  if (tid < 16) {
    int p = p0 + tid;
    float rx = ref[((size_t)bh * NP + p) * 2 + 0];
    float ry = ref[((size_t)bh * NP + p) * 2 + 1];
    float ox = offb[(b * 16 + head * 2 + 0) * NP + p];
    float oy = offb[(b * 16 + head * 2 + 1) * NP + p];
    float lx = fminf(fmaxf(rx + ox, -1.f), 1.f);
    float ly = fminf(fmaxf(ry + oy, -1.f), 1.f);
    sLoc[tid][0] = (lx + 1.f) * (WW * 0.5f) - 0.5f;
    sLoc[tid][1] = (ly + 1.f) * (HH * 0.5f) - 0.5f;
  }
  __syncthreads();

  {
    int px = tid & 15, dg = tid >> 4;
    float ix = sLoc[px][0], iy = sLoc[px][1];
    float x0f = floorf(ix), y0f = floorf(iy);
    int x0 = (int)x0f, y0 = (int)y0f;
    float wx = ix - x0f, wy = iy - y0f;
    const float* img = xin + ((size_t)b * DIM + head * DPH) * QP;
    bool i00 = (x0 >= 0) & (x0 < WW) & (y0 >= 0) & (y0 < HH);
    bool i10 = (x0 + 1 < WW) & (x0 + 1 >= 0) & (y0 >= 0) & (y0 < HH);
    bool i01 = (x0 >= 0) & (x0 < WW) & (y0 + 1 >= 0) & (y0 + 1 < HH);
    bool i11 = (x0 + 1 < WW) & (x0 + 1 >= 0) & (y0 + 1 >= 0) & (y0 + 1 < HH);
#pragma unroll
    for (int i = 0; i < 8; ++i) {
      int d = dg * 8 + i;
      const float* ch = img + (size_t)d * QP;
      float v00 = i00 ? ch[y0 * WW + x0] : 0.f;
      float v10 = i10 ? ch[y0 * WW + x0 + 1] : 0.f;
      float v01 = i01 ? ch[(y0 + 1) * WW + x0] : 0.f;
      float v11 = i11 ? ch[(y0 + 1) * WW + x0 + 1] : 0.f;
      float sv = v00 * (1.f - wy) * (1.f - wx) + v10 * (1.f - wy) * wx +
                 v01 * wy * (1.f - wx) + v11 * wy * wx;
      sS[px * 32 + d] = (_Float16)sv;
    }
  }
  __syncthreads();

  const int wave = tid >> 5, lane = tid & 31;
  const _Float16* Wm  = wave ? sWv : sWk;
  const float*    bia = wave ? v_b : k_b;
  v16h bf = load_bt(sS, 32);
  v16h a0 = load_a(Wm, 32);
  v16h a1 = load_a(Wm + 16 * 32, 32);
  v8f c0 = {}, c1 = {};
  c0 = wmma32(a0, bf, c0);
  c1 = wmma32(a1, bf, c1);

  const int col = lane & 15;
  const int rb  = (lane < 16) ? 0 : 8;
#pragma unroll
  for (int j = 0; j < 8; ++j) {
    int m = rb + j, p = p0 + col;
    float v0 = c0[j] + bia[m];
    float v1 = c1[j] + bia[16 + m];
    if (wave == 0) {
      kt[((size_t)bh * KPAD + p) * DPH + m]      = (_Float16)v0;
      kt[((size_t)bh * KPAD + p) * DPH + 16 + m] = (_Float16)v1;
    } else {
      vd[((size_t)bh * DPH + m) * KPAD + p]      = (_Float16)v0;
      vd[((size_t)bh * DPH + 16 + m) * KPAD + p] = (_Float16)v1;
    }
  }
}

// ---------------------------------------------------------------------------
// K5: flash attention per (bh, 64-q tile), 4 waves x 16 q rows.
//     Ping-pong K/V staging via global_load_async_to_lds_b128.
//     O written CHANNEL-LAST: ao[b][pix][256].
// ---------------------------------------------------------------------------
__global__ void __launch_bounds__(128)
attn_kernel(const _Float16* __restrict__ q_t, const _Float16* __restrict__ kt,
            const _Float16* __restrict__ vd, _Float16* __restrict__ ao) {
  __shared__ __align__(16) _Float16 sK[2][64 * 32];  // [kc][d]
  __shared__ __align__(16) _Float16 sV[2][32 * 64];  // [d][kc]
  __shared__ __align__(16) _Float16 sP[4][16 * 64];  // per-wave P (A layout)

  const int bh   = blockIdx.y;
  const int q0   = blockIdx.x * 64;
  const int tid  = threadIdx.x;
  const int wave = tid >> 5;
  const int lane = tid & 31;
  const int col  = lane & 15;
  const int rb   = (lane < 16) ? 0 : 8;
  const int qr   = q0 + wave * 16;

  const int krow = tid >> 1, khalf = tid & 1;   // sK: 64 rows x 2 halves
  const int drow = tid >> 2, dqrt = tid & 3;    // sV: 32 rows x 4 quarters
  uint32_t lk[2], lv[2];
  lk[0] = (uint32_t)(uintptr_t)&sK[0][0] + (uint32_t)(krow * 32 + khalf * 16) * 2;
  lk[1] = (uint32_t)(uintptr_t)&sK[1][0] + (uint32_t)(krow * 32 + khalf * 16) * 2;
  lv[0] = (uint32_t)(uintptr_t)&sV[0][0] + (uint32_t)(drow * 64 + dqrt * 16) * 2;
  lv[1] = (uint32_t)(uintptr_t)&sV[1][0] + (uint32_t)(drow * 64 + dqrt * 16) * 2;
  const _Float16* gk0 = kt + ((size_t)bh * KPAD + krow) * DPH + khalf * 16;
  const _Float16* gv0 = vd + ((size_t)bh * DPH + drow) * KPAD + dqrt * 16;

  v16h qfrag = load_a(q_t + ((size_t)bh * QP + qr) * DPH, DPH);

  float mrow[8], lrow[8];
  v8f o0 = {}, o1 = {};
#pragma unroll
  for (int j = 0; j < 8; ++j) { mrow[j] = -1e30f; lrow[j] = 0.f; }

  // prologue: stage chunk 0 into buffer 0 (4 async b128 per thread)
  async_b128(lk[0], gk0);
  async_b128(lk[0] + 16, gk0 + 8);
  async_b128(lv[0], gv0);
  async_b128(lv[0] + 16, gv0 + 8);

  for (int c = 0; c < 13; ++c) {
    const int buf = c & 1;
    if (c + 1 < 13) {   // prefetch next chunk into other buffer
      const int nb = (c + 1) & 1;
      const _Float16* gk = gk0 + (size_t)(c + 1) * 64 * DPH;
      const _Float16* gv = gv0 + (c + 1) * 64;
      async_b128(lk[nb], gk);
      async_b128(lk[nb] + 16, gk + 8);
      async_b128(lv[nb], gv);
      async_b128(lv[nb] + 16, gv + 8);
      wait_async4();   // in-order completion -> chunk c landed
    } else {
      wait_async0();
    }
    __syncthreads();

    const int kb = c * 64;
    v8f s[4];
#pragma unroll
    for (int nt = 0; nt < 4; ++nt) {
      v8f z = {};
      v16h bf = load_bt(&sK[buf][0] + nt * 16 * 32, 32);
      s[nt] = wmma32(qfrag, bf, z);
    }
#pragma unroll
    for (int nt = 0; nt < 4; ++nt) {
      bool ok = (kb + nt * 16 + col) < NP;
#pragma unroll
      for (int j = 0; j < 8; ++j)
        s[nt][j] = ok ? s[nt][j] * (1.f / 16.f) : -1e30f;
    }
    // online softmax (rows live in 16-lane half groups)
#pragma unroll
    for (int j = 0; j < 8; ++j) {
      float mj = -1e30f;
#pragma unroll
      for (int nt = 0; nt < 4; ++nt) mj = fmaxf(mj, s[nt][j]);
#pragma unroll
      for (int off = 8; off > 0; off >>= 1)
        mj = fmaxf(mj, __shfl_xor(mj, off, 32));
      float mn = fmaxf(mrow[j], mj);
      float fs = __expf(mrow[j] - mn);
      mrow[j] = mn;
      float rs = 0.f;
#pragma unroll
      for (int nt = 0; nt < 4; ++nt) {
        float p = __expf(s[nt][j] - mn);
        s[nt][j] = p;
        rs += p;
      }
#pragma unroll
      for (int off = 8; off > 0; off >>= 1) rs += __shfl_xor(rs, off, 32);
      lrow[j] = lrow[j] * fs + rs;
      o0[j] *= fs;
      o1[j] *= fs;
    }
    // C-layout -> A-layout transpose of P through wave-private LDS.
    // Same-wave LDS ops are in-order (no block barrier needed).
    _Float16* pb = &sP[wave][0];
#pragma unroll
    for (int nt = 0; nt < 4; ++nt)
#pragma unroll
      for (int j = 0; j < 8; ++j)
        pb[(rb + j) * 64 + nt * 16 + col] = (_Float16)s[nt][j];
#pragma unroll
    for (int ks = 0; ks < 2; ++ks) {
      v16h ap  = load_a(pb + ks * 32, 64);
      v16h bv0 = load_bt(&sV[buf][0] + 0 * 16 * 64 + ks * 32, 64);
      v16h bv1 = load_bt(&sV[buf][0] + 1 * 16 * 64 + ks * 32, 64);
      o0 = wmma32(ap, bv0, o0);
      o1 = wmma32(ap, bv1, o1);
    }
    __syncthreads();   // buffer `buf` free for restage at c+2
  }

  const int b = bh >> 3, head = bh & 7;
#pragma unroll
  for (int j = 0; j < 8; ++j) {
    float inv = 1.f / lrow[j];
    int qpix = qr + rb + j;
    ao[((size_t)b * QP + qpix) * DIM + head * DPH + col]      =
        (_Float16)(o0[j] * inv);
    ao[((size_t)b * QP + qpix) * DIM + head * DPH + 16 + col] =
        (_Float16)(o1[j] * inv);
  }
}

// ---------------------------------------------------------------------------
// K6: output projection: out = o_w @ ao + o_b (f32 result).
//     Entire 64x256 B panel staged once with async b128 (32 KB LDS).
// ---------------------------------------------------------------------------
__global__ void __launch_bounds__(256)
oproj_kernel(const _Float16* __restrict__ ao, const _Float16* __restrict__ owh,
             const float* __restrict__ o_b, float* __restrict__ out) {
  __shared__ __align__(16) _Float16 sB[64 * 256];   // [64 px][256 ch]

  const int b    = blockIdx.y;
  const int n0   = blockIdx.x * 64;
  const int wave = threadIdx.x >> 5;
  const int lane = threadIdx.x & 31;

  const int sj = threadIdx.x >> 2;   // pixel 0..63
  const int sq = threadIdx.x & 3;    // 8-channel quarter within 32
  const _Float16* gpix = ao + ((size_t)b * QP + n0 + sj) * DIM + sq * 8;
  const uint32_t ldst =
      (uint32_t)(uintptr_t)&sB[0] + (uint32_t)(sj * 256 + sq * 8) * 2;

#pragma unroll
  for (int kc = 0; kc < 8; ++kc)
    async_b128(ldst + kc * 64, gpix + kc * 32);

  v8f acc[8];
#pragma unroll
  for (int i = 0; i < 8; ++i) acc[i] = (v8f){};

  wait_async0();
  __syncthreads();

  for (int kc = 0; kc < 8; ++kc) {
    v16h a0 = load_a(owh + (wave * 32 + 0) * DIM + kc * 32, DIM);
    v16h a1 = load_a(owh + (wave * 32 + 16) * DIM + kc * 32, DIM);
#pragma unroll
    for (int nt = 0; nt < 4; ++nt) {
      v16h bf = load_bt(sB + nt * 16 * 256 + kc * 32, 256);
      acc[nt]     = wmma32(a0, bf, acc[nt]);
      acc[4 + nt] = wmma32(a1, bf, acc[4 + nt]);
    }
  }

  const int col = lane & 15;
  const int rb  = (lane < 16) ? 0 : 8;
#pragma unroll
  for (int mt = 0; mt < 2; ++mt)
#pragma unroll
    for (int nt = 0; nt < 4; ++nt)
#pragma unroll
      for (int j = 0; j < 8; ++j) {
        int ch  = (wave * 2 + mt) * 16 + rb + j;
        int pix = n0 + nt * 16 + col;
        out[((size_t)b * DIM + ch) * QP + pix] = acc[mt * 4 + nt][j] + o_b[ch];
      }
}

// ---------------------------------------------------------------------------
extern "C" void kernel_launch(void* const* d_in, const int* in_sizes, int n_in,
                              void* d_out, int out_size, void* d_ws,
                              size_t ws_size, hipStream_t stream) {
  const float* query = (const float*)d_in[0];
  const float* xin   = (const float*)d_in[1];
  const float* refp  = (const float*)d_in[2];
  const float* pos   = (const float*)d_in[3];
  const float* so_w1 = (const float*)d_in[4];
  const float* so_b1 = (const float*)d_in[5];
  const float* ln_g  = (const float*)d_in[6];
  const float* ln_b  = (const float*)d_in[7];
  const float* so_w2 = (const float*)d_in[8];
  const float* so_b2 = (const float*)d_in[9];
  const float* q_w   = (const float*)d_in[10];
  const float* q_b   = (const float*)d_in[11];
  const float* k_w   = (const float*)d_in[12];
  const float* k_b   = (const float*)d_in[13];
  const float* v_w   = (const float*)d_in[14];
  const float* v_b   = (const float*)d_in[15];
  const float* o_w   = (const float*)d_in[16];
  const float* o_b   = (const float*)d_in[17];
  float* out = (float*)d_out;

  // workspace carve
  char* w = (char*)d_ws;
  _Float16* recP = (_Float16*)w; w += (size_t)BATCH * 512 * 32 * 32 * 2;
  float*    offb = (float*)w;    w += (size_t)BATCH * 16 * NP * 4;
  _Float16* q_t  = (_Float16*)w; w += (size_t)32 * QP * DPH * 2;
  _Float16* kt   = (_Float16*)w; w += (size_t)32 * KPAD * DPH * 2;
  _Float16* vd   = (_Float16*)w; w += (size_t)32 * DPH * KPAD * 2;
  _Float16* ao   = (_Float16*)w; w += (size_t)BATCH * QP * DIM * 2;
  _Float16* w1h  = (_Float16*)w; w += (size_t)W1ELEMS * 2;
  _Float16* qwh  = (_Float16*)w; w += (size_t)65536 * 2;
  _Float16* owh  = (_Float16*)w; w += (size_t)65536 * 2;

  convert_weights<<<(W1ELEMS + 131072 + 255) / 256, 256, 0, stream>>>(
      so_w1, q_w, o_w, w1h, qwh, owh);
  pool_pad_kernel<<<(BATCH * 512 * 32 * 32) / 256, 256, 0, stream>>>(
      query, xin, recP);
  offsets_kernel<<<dim3(49, BATCH), 256, 0, stream>>>(
      recP, w1h, so_b1, ln_g, ln_b, so_w2, so_b2, offb);
  qproj_kernel<<<dim3(49, BATCH), 256, 0, stream>>>(query, pos, qwh, q_b, q_t);
  sample_kv_kernel<<<dim3(49, 32), 64, 0, stream>>>(
      xin, refp, offb, k_w, k_b, v_w, v_b, kt, vd);
  attn_kernel<<<dim3(49, 32), 128, 0, stream>>>(q_t, kt, vd, ao);
  oproj_kernel<<<dim3(49, BATCH), 256, 0, stream>>>(ao, owh, o_b, out);
}
